// SRP_66949950210443
// MI455X (gfx1250) — compile-verified, matrix operand order checked by reading, not run on hardware
//
#include <hip/hip_runtime.h>
#include <stdint.h>

typedef __attribute__((ext_vector_type(16))) _Float16 v16h;
typedef __attribute__((ext_vector_type(8)))  float    v8f;

#define N_SAMPLES 2400000
#define N_MICS    8
#define NFFT      512
#define HOP       256
#define NFRAMES   9376          // 1 + N_SAMPLES/HOP
#define NTILE_T   586           // NFRAMES/16
#define F0BIN     13
#define NF        19            // bins 13..31
#define NJ        361
#define JTILES    46            // ceil(361/8)
#define JPAD      368
#define NSPLIT    8
#define TCHUNK    74            // ceil(586/8)
#define PI_F      3.14159265358979323846f
#define TWO_PI_F  6.28318530717958647692f

// ---------------- workspace layout (bytes) ----------------
#define OFF_BASIS 0
#define SZ_BASIS  (16*3*32*8*4)                 // 49,152
#define OFF_STEER (OFF_BASIS + SZ_BASIS)
#define SZ_STEER  (NF*JTILES*32*8*4)            // 894,976
#define OFF_SIG   (OFF_STEER + SZ_STEER)        // 944,128 (16B aligned)
#define SZ_SIG    (NF*NFRAMES*16*2)             // 5,700,608
#define OFF_PART  (OFF_SIG + SZ_SIG)            // 6,644,736
#define SZ_PART   (NSPLIT*NF*JPAD*4)            // 223,744

// ---- CDNA5 async global->LDS copy (ASYNCcnt path), guarded so it can't regress ----
#if defined(__has_builtin)
#if __has_builtin(__builtin_amdgcn_global_load_async_to_lds_b32)
#define HAVE_ASYNC_LDS 1
#endif
#endif

#ifdef HAVE_ASYNC_LDS
typedef __attribute__((address_space(1))) int* as1_i32;
typedef __attribute__((address_space(3))) int* as3_i32;
__device__ __forceinline__ void async_wait0() {
#if __has_builtin(__builtin_amdgcn_s_wait_asynccnt)
    __builtin_amdgcn_s_wait_asynccnt(0);
#else
    asm volatile("s_wait_asynccnt 0x0" ::: "memory");
#endif
}
#endif

__device__ __forceinline__ uint32_t pack_h2(float lo, float hi) {
    union { _Float16 h[2]; uint32_t u; } r;
    r.h[0] = (_Float16)lo; r.h[1] = (_Float16)hi;
    return r.u;
}

// 16-bit B-matrix (32x16) fragment: per lane, VGPR v holds K pair:
//   lanes 0-15 : v<4 -> K={2v,2v+1}        ; v>=4 -> K={16+2(v-4), ...}
//   lanes 16-31: v<4 -> K={8+2v,8+2v+1}    ; v>=4 -> K={24+2(v-4), ...}
// (mirror of the documented 16-bit A 16x32 layout)
__device__ __forceinline__ int kpair_base(int lane, int v) {
    int kb = (lane < 16) ? 0 : 8;
    return (v < 4) ? (kb + 2 * v) : (kb + 16 + 2 * (v - 4));
}

// ---------------- init: DFT basis fragments ----------------
// basis column co: co<38 -> bin fi=co>>1; even=cos(ang), odd=-sin(ang); co>=38 -> 0
__global__ __launch_bounds__(32) void srp_init_basis(uint32_t* __restrict__ basis) {
    int lane = threadIdx.x;
    int kstep = blockIdx.x;          // 0..15 (K chunk of 32 over n=0..511)
    int ct    = blockIdx.y;          // 0..2  (column tile)
    int ncol  = ct * 16 + (lane & 15);
    uint32_t* dst = basis + (((size_t)kstep * 3 + ct) * 32 + lane) * 8;
    for (int v = 0; v < 8; ++v) {
        int kk = kpair_base(lane, v);
        float lo = 0.0f, hi = 0.0f;
        if (ncol < 2 * NF) {
            int   kbin = F0BIN + (ncol >> 1);
            float w    = TWO_PI_F * (float)kbin / (float)NFFT;
            int   n0   = kstep * 32 + kk;
            float a0 = w * (float)n0, a1 = w * (float)(n0 + 1);
            if (ncol & 1) { lo = -sinf(a0); hi = -sinf(a1); }
            else          { lo =  cosf(a0); hi =  cosf(a1); }
        }
        dst[v] = pack_h2(lo, hi);
    }
}

// ---------------- init: steering fragments ----------------
// A2 row (stage 2): K=0..7 -> sig_re[c], K=8..15 -> sig_im[c], K>=16 -> 0
// B2[k][n]: n even (re out): k<8 -> cos(ph), k in 8..15 -> sin(ph)
//           n odd  (im out): k<8 -> -sin(ph), k in 8..15 -> cos(ph)
// ph = 2*pi*d*f*c*sin(theta_j)/343
__global__ __launch_bounds__(32) void srp_init_steer(uint32_t* __restrict__ steer) {
    int lane = threadIdx.x;
    int fi = blockIdx.x;             // 0..18
    int jt = blockIdx.y;             // 0..45
    int ncol = lane & 15;
    int j = jt * 8 + (ncol >> 1);
    uint32_t* dst = steer + (((size_t)fi * JTILES + jt) * 32 + lane) * 8;
    float fhz   = (float)(F0BIN + fi) * (16000.0f / 512.0f);
    float theta = (-90.0f + 0.5f * (float)j) * (PI_F / 180.0f);
    float coef  = TWO_PI_F * 0.04f * fhz * sinf(theta) / 343.0f;
    for (int v = 0; v < 8; ++v) {
        int kk = kpair_base(lane, v);
        float e[2] = {0.0f, 0.0f};
        for (int t = 0; t < 2; ++t) {
            int k = kk + t;
            if (k >= 16 || j >= NJ) continue;
            int   c  = k & 7;
            float ph = coef * (float)c;
            if ((ncol & 1) == 0) e[t] = (k < 8) ?  cosf(ph) : sinf(ph);
            else                 e[t] = (k < 8) ? -sinf(ph) : cosf(ph);
        }
        dst[v] = pack_h2(e[0], e[1]);
    }
}

// ---------------- stage 1: STFT (19 bins) via WMMA ----------------
// One wave per (mic, 16-frame tile). A = windowed frames (f16), B = DFT basis.
__global__ __launch_bounds__(32) void srp_stft(const float* __restrict__ x,
                                               const uint32_t* __restrict__ basis,
                                               _Float16* __restrict__ sig) {
    __shared__ float sx[(16 - 1) * HOP + NFFT];   // 4352 samples (overlapping frames)
    __shared__ float sw[NFFT];                    // sqrt-hann = sin(pi*n/NFFT)
    int lane = threadIdx.x;
    int c  = blockIdx.x;             // mic
    int tt = blockIdx.y;             // frame tile
    long start = (long)tt * 16 * HOP - 256;       // reflect-padded start
    for (int i = lane; i < 4352; i += 32) {
        long s = start + i;
        if (s < 0) s = -s;                               // reflect left
        if (s >= N_SAMPLES) s = 2L * N_SAMPLES - 2 - s;  // reflect right
#ifdef HAVE_ASYNC_LDS
        // CDNA5 async copy: memory -> LDS without a VGPR round-trip (ASYNCcnt)
        __builtin_amdgcn_global_load_async_to_lds_b32(
            (as1_i32)(x + s * N_MICS + c), (as3_i32)&sx[i], 0, 0);
#else
        sx[i] = x[s * N_MICS + c];
#endif
    }
    for (int i = lane; i < NFFT; i += 32)
        sw[i] = sinf((float)i * (PI_F / (float)NFFT));
#ifdef HAVE_ASYNC_LDS
    async_wait0();
#endif
    __syncthreads();

    v8f acc[3] = {};
    int m  = lane & 15;
    int kb = (lane < 16) ? 0 : 8;    // 16-bit A 16x32 K-half per lane group
    const float* row = &sx[m * HOP];
    for (int kstep = 0; kstep < 16; ++kstep) {
        v16h a;
        int n0 = kstep * 32 + kb;
#pragma unroll
        for (int e = 0; e < 8; ++e) { int n = n0 + e;      a[e]     = (_Float16)(row[n] * sw[n]); }
#pragma unroll
        for (int e = 0; e < 8; ++e) { int n = n0 + 16 + e; a[8 + e] = (_Float16)(row[n] * sw[n]); }
#pragma unroll
        for (int ct = 0; ct < 3; ++ct) {
            const uint4* bp = (const uint4*)(basis + (((size_t)kstep * 3 + ct) * 32 + lane) * 8);
            union { uint4 u[2]; v16h h; } B;
            B.u[0] = bp[0]; B.u[1] = bp[1];
            acc[ct] = __builtin_amdgcn_wmma_f32_16x16x32_f16(
                false, a, false, B.h, (short)0, acc[ct], false, false);
        }
    }
    // D layout: lane l, VGPR v -> row m = v + (l<16?0:8), col n = l&15
    int mrow = (lane < 16) ? 0 : 8;
    int t0   = tt * 16;
    for (int ct = 0; ct < 3; ++ct) {
        int co = ct * 16 + (lane & 15);
        if (co >= 2 * NF) continue;
        int fi = co >> 1, isIm = co & 1;
#pragma unroll
        for (int v = 0; v < 8; ++v) {
            int t = t0 + mrow + v;
            sig[((size_t)fi * NFRAMES + t) * 16 + c + isIm * 8] = (_Float16)acc[ct][v];
        }
    }
}

// ---------------- stage 2: beamform + |.| + frame reduction (fused) ----------------
// One wave per (j-tile, freq, t-split). One WMMA per 16-frame tile.
__global__ __launch_bounds__(32) void srp_beam(const _Float16* __restrict__ sig,
                                               const uint32_t* __restrict__ steer,
                                               float* __restrict__ part) {
    int lane = threadIdx.x;
    int jt = blockIdx.x, fi = blockIdx.y, sp = blockIdx.z;
    union { uint4 u[2]; v16h h; } B;
    const uint4* bp = (const uint4*)(steer + (((size_t)fi * JTILES + jt) * 32 + lane) * 8);
    B.u[0] = bp[0]; B.u[1] = bp[1];

    int halfsel = (lane < 16) ? 0 : 8;   // re half (K=0..7) vs im half (K=8..15)
    int mlane   = lane & 15;
    float accm  = 0.0f;
    int t_begin = sp * TCHUNK;
    int t_end   = t_begin + TCHUNK; if (t_end > NTILE_T) t_end = NTILE_T;

    for (int tt = t_begin; tt < t_end; ++tt) {
        int tR = tt * 16 + mlane;
        const uint4* ap = (const uint4*)(sig + ((size_t)fi * NFRAMES + tR) * 16 + halfsel);
        if (tt + 1 < t_end)
            __builtin_prefetch((const void*)(ap + 32), 0, 0);   // next 16-frame tile (+512B)
        union { uint4 u; _Float16 h[8]; } A8; A8.u = *ap;
        v16h a = {};                      // K=16..31 zero-padded
#pragma unroll
        for (int e = 0; e < 8; ++e) a[e] = A8.h[e];
        v8f z = {};
        v8f d = __builtin_amdgcn_wmma_f32_16x16x32_f16(
            false, a, false, B.h, (short)0, z, false, false);
        // cols: even = re, odd = im -> pair lanes via shfl_xor(1)
#pragma unroll
        for (int v = 0; v < 8; ++v) {
            float other = __shfl_xor(d[v], 1, 32);
            accm += sqrtf(d[v] * d[v] + other * other);
        }
    }
    // lanes l and l^16 cover the two row halves of the same column
    accm += __shfl_xor(accm, 16, 32);
    if (lane < 16 && (lane & 1) == 0) {
        int j = jt * 8 + (lane >> 1);
        if (j < NJ)
            part[((size_t)sp * NF + fi) * JPAD + j] = accm;     // deterministic (no atomics)
    }
}

// ---------------- stage 3: power map + argmax ----------------
__global__ __launch_bounds__(512) void srp_final(const float* __restrict__ part,
                                                 float* __restrict__ out) {
    __shared__ float sval[512];
    __shared__ int   sidx[512];
    int j = threadIdx.x;
    float p = -1.0f;
    if (j < NJ) {
        p = 0.0f;
        for (int fi = 0; fi < NF; ++fi) {
            float ssum = 0.0f;
            for (int sp = 0; sp < NSPLIT; ++sp)
                ssum += part[((size_t)sp * NF + fi) * JPAD + j];
            float ma = ssum / (float)NFRAMES;   // mean |resp| over frames
            p += ma * ma;
        }
        p /= (float)NF;
        out[1 + j] = p;
    }
    sval[j] = p; sidx[j] = j;
    __syncthreads();
    for (int off = 256; off > 0; off >>= 1) {
        if (j < off) {
            bool take = (sval[j + off] > sval[j]) ||
                        (sval[j + off] == sval[j] && sidx[j + off] < sidx[j]);
            if (take) { sval[j] = sval[j + off]; sidx[j] = sidx[j + off]; }
        }
        __syncthreads();
    }
    if (j == 0) out[0] = (float)sidx[0] * 0.5f - 90.0f;
}

extern "C" void kernel_launch(void* const* d_in, const int* in_sizes, int n_in,
                              void* d_out, int out_size, void* d_ws, size_t ws_size,
                              hipStream_t stream) {
    (void)in_sizes; (void)n_in; (void)out_size; (void)ws_size;
    const float* x = (const float*)d_in[0];
    char* ws = (char*)d_ws;
    uint32_t* basis = (uint32_t*)(ws + OFF_BASIS);
    uint32_t* steer = (uint32_t*)(ws + OFF_STEER);
    _Float16* sig   = (_Float16*)(ws + OFF_SIG);
    float*    part  = (float*)(ws + OFF_PART);
    float*    out   = (float*)d_out;

    srp_init_basis<<<dim3(16, 3), 32, 0, stream>>>(basis);
    srp_init_steer<<<dim3(NF, JTILES), 32, 0, stream>>>(steer);
    srp_stft<<<dim3(N_MICS, NTILE_T), 32, 0, stream>>>(x, basis, sig);
    srp_beam<<<dim3(JTILES, NF, NSPLIT), 32, 0, stream>>>(sig, steer, part);
    srp_final<<<1, 512, 0, stream>>>(part, out);
}
